// LSTM_Predictor_48576080118527
// MI455X (gfx1250) — matrix operand order repaired and checked by reading, not compiled, not run on hardware
//
#include <hip/hip_runtime.h>

typedef __attribute__((ext_vector_type(16))) _Float16 v16h;
typedef __attribute__((ext_vector_type(8)))  _Float16 v8h;
typedef __attribute__((ext_vector_type(8)))  float    v8f;
typedef __attribute__((ext_vector_type(2)))  float    v2f;

#define B_TOTAL 4096
#define T_STEPS 512
#define IN_DIM  4
#define HDIM    64
#define G4      256                      // 4*H gate columns
#define WAVES_PER_BLOCK 2
#define THREADS (WAVES_PER_BLOCK * 32)
#define TILES_PER_WAVE 2                 // two 16-row tiles share every B fragment
#define ROWS_PER_WAVE (16 * TILES_PER_WAVE)              // 32
#define ROWS_PER_BLOCK (WAVES_PER_BLOCK * ROWS_PER_WAVE) // 64

#if defined(__gfx1250__) && __has_builtin(__builtin_amdgcn_wmma_f32_16x16x4_f32)
#define HAVE_WMMA4 1
#else
#define HAVE_WMMA4 0
#endif

#if defined(__gfx1250__) && __has_builtin(__builtin_amdgcn_tanhf)
__device__ __forceinline__ float ftanh_f(float x) { return __builtin_amdgcn_tanhf(x); }
#else
__device__ __forceinline__ float ftanh_f(float x) {
  return 1.0f - 2.0f * __builtin_amdgcn_rcpf(__expf(2.0f * x) + 1.0f);
}
#endif
// sigmoid(acc + b) with pre-scaled hb = 0.5*b:  fma -> v_tanh -> fma (1 trans op)
__device__ __forceinline__ float fsig_pre(float acc, float hb) {
  return fmaf(0.5f, ftanh_f(fmaf(0.5f, acc, hb)), 0.5f);
}

// A fragment (16x32 f16) from plain row-major h[16][64] in LDS.
// Lane L: row m=L&15; elem j<8 -> K = ks*32 + (L/16)*8 + j ; j>=8 -> +16.
__device__ __forceinline__ v16h afrag(const _Float16* h, int L, int ks) {
  const int base = (L & 15) * HDIM + ks * 32 + ((L >> 4) & 1) * 8;
  v8h lo = *(const v8h*)(h + base);
  v8h hi = *(const v8h*)(h + base + 16);
  return __builtin_shufflevector(lo, hi, 0,1,2,3,4,5,6,7,8,9,10,11,12,13,14,15);
}

// B fragment (32x16 f16): pre-swizzled contiguous 32B per lane.
__device__ __forceinline__ v16h bfrag(const _Float16* w, int tile, int ks, int L) {
  return *(const v16h*)(w + (((tile * 2 + ks) * 32) + L) * 16);
}

__device__ __forceinline__ v8f wmma16(v16h a, v16h b, v8f c) {
  return __builtin_amdgcn_wmma_f32_16x16x32_f16(false, a, false, b, (short)0, c,
                                                false, false);
}

__global__ __launch_bounds__(THREADS)
void lstm2_wmma_kernel(const float* __restrict__ x,
                       const float* __restrict__ Wih0, const float* __restrict__ Whh0,
                       const float* __restrict__ bih0, const float* __restrict__ bhh0,
                       const float* __restrict__ Wih1, const float* __restrict__ Whh1,
                       const float* __restrict__ bih1, const float* __restrict__ bhh1,
                       const float* __restrict__ Wfc,  const float* __restrict__ bfc,
                       float* __restrict__ out) {
  extern __shared__ char smem[];
  _Float16* whh0s = (_Float16*)smem;              // 16384 halves, B-frag swizzled
  _Float16* wih1s = whh0s + G4 * HDIM;
  _Float16* whh1s = wih1s + G4 * HDIM;
  float*    w0t   = (float*)(whh1s + G4 * HDIM);  // [4][256] = Wih0^T f32
  float*    bias0 = w0t + IN_DIM * G4;            // 256 (bih0+bhh0)
  float*    bias1 = bias0 + G4;                   // 256
  _Float16* hbuf  = (_Float16*)(bias1 + G4);      // per wave: h0[32][64], h1[32][64]
  float*    xs_all = (float*)(hbuf + WAVES_PER_BLOCK * 2 * ROWS_PER_WAVE * HDIM);

  const int tid = threadIdx.x;

  // --- preload: weights f32 -> f16 swizzled into B-fragment layout ---
  for (int idx = tid; idx < G4 * HDIM; idx += THREADS) {
    const int g = idx >> 6, k = idx & 63;   // weight[g][k], row-major [256][64]
    const int off = ((((g >> 4) * 2 + (k >> 5)) * 32) +
                     (((k >> 3) & 1) * 16 + (g & 15))) * 16 + (k & 7) + ((k >> 4) & 1) * 8;
    whh0s[off] = (_Float16)Whh0[idx];
    wih1s[off] = (_Float16)Wih1[idx];
    whh1s[off] = (_Float16)Whh1[idx];
  }
  for (int idx = tid; idx < G4 * IN_DIM; idx += THREADS) {
    const int g = idx >> 2, k = idx & 3;    // Wih0[g][k], [256][4]
    w0t[k * G4 + g] = Wih0[idx];
  }
  for (int idx = tid; idx < G4; idx += THREADS) {
    bias0[idx] = bih0[idx] + bhh0[idx];
    bias1[idx] = bih1[idx] + bhh1[idx];
  }
  for (int idx = tid; idx < WAVES_PER_BLOCK * 2 * ROWS_PER_WAVE * HDIM; idx += THREADS)
    hbuf[idx] = (_Float16)0.0f;
  __syncthreads();

  const int w = tid >> 5, L = tid & 31;
  const int ln = L & 15;
  const int half = (L >> 4) & 1;
  const int rowBase = (blockIdx.x * WAVES_PER_BLOCK + w) * ROWS_PER_WAVE;
  _Float16* h0 = hbuf + w * 2 * ROWS_PER_WAVE * HDIM;   // [32][64]
  _Float16* h1 = h0 + ROWS_PER_WAVE * HDIM;             // [32][64]
  float* xs = xs_all + w * ROWS_PER_WAVE * IN_DIM;      // [32][4]

  // Hoist time-invariant per-lane operands into registers:
  //  - biases: sigmoid gates (q!=2) pre-scaled by 0.5, tanh gate raw
  //  - wmma4 B operands (Wih0^T columns)
  float hb0[16], hb1[16];
#pragma unroll
  for (int tile = 0; tile < 16; ++tile) {
    const int q = tile >> 2;
    const float b0v = bias0[tile * 16 + ln];
    const float b1v = bias1[tile * 16 + ln];
    hb0[tile] = (q == 2) ? b0v : 0.5f * b0v;
    hb1[tile] = (q == 2) ? b1v : 0.5f * b1v;
  }
#if HAVE_WMMA4
  v2f bxr[16];
#pragma unroll
  for (int tile = 0; tile < 16; ++tile) {
    const int col = tile * 16 + ln;
    bxr[tile].x = w0t[(half * 2 + 0) * G4 + col];
    bxr[tile].y = w0t[(half * 2 + 1) * G4 + col];
  }
#endif

  float c0r[TILES_PER_WAVE][4][8], c1r[TILES_PER_WAVE][4][8];
#pragma unroll
  for (int p = 0; p < TILES_PER_WAVE; ++p)
#pragma unroll
    for (int j = 0; j < 4; ++j)
#pragma unroll
      for (int r = 0; r < 8; ++r) { c0r[p][j][r] = 0.f; c1r[p][j][r] = 0.f; }

  const v8f vzero = {};

#pragma unroll 1
  for (int t = 0; t < T_STEPS; ++t) {
    // stage x_t rows into LDS (each lane owns one of the wave's 32 batch rows)
    {
      const size_t ro = ((size_t)(rowBase + L) * T_STEPS + t) * IN_DIM;
      const float4 xv = *(const float4*)(x + ro);
      *(float4*)(xs + L * 4) = xv;
      if (t + 16 < T_STEPS)
        __builtin_prefetch(x + ro + 16 * IN_DIM, 0, 1);
    }

    // ---------------- layer 0 ----------------
    {
      v16h a0[TILES_PER_WAVE], a1[TILES_PER_WAVE];
#if HAVE_WMMA4
      v2f ax[TILES_PER_WAVE];
#endif
#pragma unroll
      for (int p = 0; p < TILES_PER_WAVE; ++p) {
        a0[p] = afrag(h0 + p * 16 * HDIM, L, 0);
        a1[p] = afrag(h0 + p * 16 * HDIM, L, 1);
#if HAVE_WMMA4
        ax[p].x = xs[(p * 16 + ln) * 4 + half * 2 + 0];
        ax[p].y = xs[(p * 16 + ln) * 4 + half * 2 + 1];
#endif
      }
#pragma unroll
      for (int j = 0; j < 4; ++j) {
        v8f acc[TILES_PER_WAVE][4];
#pragma unroll
        for (int q = 0; q < 4; ++q) {           // gates i,f,g,o
          const int tile = q * 4 + j;
#if HAVE_WMMA4
          acc[0][q] = __builtin_amdgcn_wmma_f32_16x16x4_f32(
              false, ax[0], false, bxr[tile], (short)0, vzero, false, false);
          acc[1][q] = __builtin_amdgcn_wmma_f32_16x16x4_f32(
              false, ax[1], false, bxr[tile], (short)0, vzero, false, false);
#else
          const int col = tile * 16 + ln;
#pragma unroll
          for (int p = 0; p < TILES_PER_WAVE; ++p) {
#pragma unroll
            for (int r = 0; r < 8; ++r) {
              const int m = p * 16 + (half ? r + 8 : r);
              float s = 0.f;
#pragma unroll
              for (int k = 0; k < IN_DIM; ++k) s += xs[m * 4 + k] * w0t[k * G4 + col];
              acc[p][q][r] = s;
            }
          }
#endif
          const v16h b0 = bfrag(whh0s, tile, 0, L);   // shared by both tiles
          acc[0][q] = wmma16(a0[0], b0, acc[0][q]);
          acc[1][q] = wmma16(a0[1], b0, acc[1][q]);
          const v16h b1 = bfrag(whh0s, tile, 1, L);
          acc[0][q] = wmma16(a1[0], b1, acc[0][q]);
          acc[1][q] = wmma16(a1[1], b1, acc[1][q]);
        }
#pragma unroll
        for (int p = 0; p < TILES_PER_WAVE; ++p)
#pragma unroll
          for (int r = 0; r < 8; ++r) {
            const float iv = fsig_pre(acc[p][0][r], hb0[0 * 4 + j]);
            const float fv = fsig_pre(acc[p][1][r], hb0[1 * 4 + j]);
            const float gv = ftanh_f(acc[p][2][r] + hb0[2 * 4 + j]);
            const float ov = fsig_pre(acc[p][3][r], hb0[3 * 4 + j]);
            const float cn = fv * c0r[p][j][r] + iv * gv;
            c0r[p][j][r] = cn;
            const int m = p * 16 + (half ? r + 8 : r);
            h0[m * HDIM + j * 16 + ln] = (_Float16)(ov * ftanh_f(cn));
          }
      }
    }

    // ---------------- layer 1 ----------------
    {
      v16h an0[TILES_PER_WAVE], an1[TILES_PER_WAVE];
      v16h ar0[TILES_PER_WAVE], ar1[TILES_PER_WAVE];
#pragma unroll
      for (int p = 0; p < TILES_PER_WAVE; ++p) {
        an0[p] = afrag(h0 + p * 16 * HDIM, L, 0);   // new h0 = layer-1 input
        an1[p] = afrag(h0 + p * 16 * HDIM, L, 1);
        ar0[p] = afrag(h1 + p * 16 * HDIM, L, 0);   // old h1 = recurrent input
        ar1[p] = afrag(h1 + p * 16 * HDIM, L, 1);
      }
#pragma unroll
      for (int j = 0; j < 4; ++j) {
        v8f acc[TILES_PER_WAVE][4];
#pragma unroll
        for (int q = 0; q < 4; ++q) {
          const int tile = q * 4 + j;
          v16h b;
          b = bfrag(wih1s, tile, 0, L);
          acc[0][q] = wmma16(an0[0], b, vzero);
          acc[1][q] = wmma16(an0[1], b, vzero);
          b = bfrag(wih1s, tile, 1, L);
          acc[0][q] = wmma16(an1[0], b, acc[0][q]);
          acc[1][q] = wmma16(an1[1], b, acc[1][q]);
          b = bfrag(whh1s, tile, 0, L);
          acc[0][q] = wmma16(ar0[0], b, acc[0][q]);
          acc[1][q] = wmma16(ar0[1], b, acc[1][q]);
          b = bfrag(whh1s, tile, 1, L);
          acc[0][q] = wmma16(ar1[0], b, acc[0][q]);
          acc[1][q] = wmma16(ar1[1], b, acc[1][q]);
        }
#pragma unroll
        for (int p = 0; p < TILES_PER_WAVE; ++p)
#pragma unroll
          for (int r = 0; r < 8; ++r) {
            const float iv = fsig_pre(acc[p][0][r], hb1[0 * 4 + j]);
            const float fv = fsig_pre(acc[p][1][r], hb1[1 * 4 + j]);
            const float gv = ftanh_f(acc[p][2][r] + hb1[2 * 4 + j]);
            const float ov = fsig_pre(acc[p][3][r], hb1[3 * 4 + j]);
            const float cn = fv * c1r[p][j][r] + iv * gv;
            c1r[p][j][r] = cn;
            const int m = p * 16 + (half ? r + 8 : r);
            h1[m * HDIM + j * 16 + ln] = (_Float16)(ov * ftanh_f(cn));
          }
      }
    }
  }

  // ---------------- FC: out[b, 0:20] = h1_last @ Wfc^T + bfc ----------------
#pragma unroll 1
  for (int idx = L; idx < ROWS_PER_WAVE * 20; idx += 32) {
    const int m = idx / 20, col = idx - m * 20;
    float s = bfc[col];
#pragma unroll 1
    for (int k = 0; k < HDIM; ++k)
      s += (float)h1[m * HDIM + k] * Wfc[col * HDIM + k];
    out[(size_t)(rowBase + m) * 20 + col] = s;
  }
}

extern "C" void kernel_launch(void* const* d_in, const int* in_sizes, int n_in,
                              void* d_out, int out_size, void* d_ws, size_t ws_size,
                              hipStream_t stream) {
  (void)in_sizes; (void)n_in; (void)out_size; (void)d_ws; (void)ws_size;
  const float* x    = (const float*)d_in[0];
  const float* Wih0 = (const float*)d_in[1];
  const float* Whh0 = (const float*)d_in[2];
  const float* bih0 = (const float*)d_in[3];
  const float* bhh0 = (const float*)d_in[4];
  const float* Wih1 = (const float*)d_in[5];
  const float* Whh1 = (const float*)d_in[6];
  const float* bih1 = (const float*)d_in[7];
  const float* bhh1 = (const float*)d_in[8];
  const float* Wfc  = (const float*)d_in[9];
  const float* bfc  = (const float*)d_in[10];
  float* out = (float*)d_out;

  const size_t lds = 3 * (size_t)(G4 * HDIM) * 2                       // f16 weights
                   + (size_t)(IN_DIM * G4 + 2 * G4) * 4                // w0t + biases
                   + (size_t)(WAVES_PER_BLOCK * 2 * ROWS_PER_WAVE * HDIM) * 2  // h0/h1
                   + (size_t)(WAVES_PER_BLOCK * ROWS_PER_WAVE * IN_DIM) * 4;   // xs

  hipFuncSetAttribute((const void*)lstm2_wmma_kernel,
                      hipFuncAttributeMaxDynamicSharedMemorySize, (int)lds);

  const dim3 grid(B_TOTAL / ROWS_PER_BLOCK);   // 64 blocks
  const dim3 block(THREADS);                   // 64 threads = 2 waves
  lstm2_wmma_kernel<<<grid, block, lds, stream>>>(
      x, Wih0, Whh0, bih0, bhh0, Wih1, Whh1, bih1, bhh1, Wfc, bfc, out);
}
